// RWKV7cTimeMix_14946486190860
// MI455X (gfx1250) — compile-verified, matrix operand order checked by reading, not run on hardware
//
#include <hip/hip_runtime.h>

#define Bn 2
#define Tn 1024
#define Cn 1024
#define Hn 16
#define Nh 64
#define Mn (Bn*Tn)
#define DW 64
#define DA 64
#define DV 32
#define DG 160

typedef __attribute__((ext_vector_type(16))) __bf16          v16bf;
typedef __attribute__((ext_vector_type(16))) unsigned short  v16u;
typedef __attribute__((ext_vector_type(8)))  unsigned short  v8u;
typedef __attribute__((ext_vector_type(8)))  float           v8f;

#if defined(__HIP_DEVICE_COMPILE__) && \
    __has_builtin(__builtin_amdgcn_global_load_async_to_lds_b32) && \
    __has_builtin(__builtin_amdgcn_s_wait_asynccnt)
#define HAVE_ASYNC_LDS 1
#else
#define HAVE_ASYNC_LDS 0
#endif

static __device__ __forceinline__ unsigned short f2bf(float f) {
  unsigned int u = __float_as_uint(f);
  u += 0x7FFFu + ((u >> 16) & 1u);         // round-to-nearest-even
  return (unsigned short)(u >> 16);
}
static __device__ __forceinline__ float sigmoidf_(float x) {
  return 1.f / (1.f + __expf(-x));
}
// two aligned 16B loads -> one 16-element bf16 WMMA fragment
static __device__ __forceinline__ v16bf ldfrag(const unsigned short* __restrict__ p,
                                               int o1, int o2) {
  v8u lo = *(const v8u*)(p + o1);
  v8u hi = *(const v8u*)(p + o2);
  v16u u = __builtin_shufflevector(lo, hi, 0,1,2,3,4,5,6,7,8,9,10,11,12,13,14,15);
  return __builtin_bit_cast(v16bf, u);
}

#if HAVE_ASYNC_LDS
static __device__ __forceinline__ void async_ld(const float* g, float* l) {
  __builtin_amdgcn_global_load_async_to_lds_b32(
      (__attribute__((address_space(1))) int*)(void*)(const_cast<float*>(g)),
      (__attribute__((address_space(3))) int*)(void*)l, 0, 0);
}
#endif

// ---------------- elementwise converts ----------------
__global__ void k_f2bf(const float* __restrict__ s, unsigned short* __restrict__ d, int n) {
  int i = blockIdx.x * blockDim.x + threadIdx.x;
  if (i < n) d[i] = f2bf(s[i]);
}
// convert + transpose: s[K,N] f32 -> d[N,K] bf16  (for WMMA B operands)
__global__ void k_f2bf_t(const float* __restrict__ s, unsigned short* __restrict__ d,
                         int K, int N) {
  int i = blockIdx.x * blockDim.x + threadIdx.x;
  if (i >= K * N) return;
  int k = i / N, n = i - k * N;
  d[(size_t)n * K + k] = f2bf(s[i]);
}
__global__ void k_tanh_bf(const float* __restrict__ s, unsigned short* __restrict__ d, int n) {
  int i = blockIdx.x * blockDim.x + threadIdx.x;
  if (i < n) d[i] = f2bf(tanhf(s[i]));
}
__global__ void k_sig_bf(const float* __restrict__ s, unsigned short* __restrict__ d, int n) {
  int i = blockIdx.x * blockDim.x + threadIdx.x;
  if (i < n) d[i] = f2bf(sigmoidf_(s[i]));
}

// ---------------- token shift + mix ----------------
__global__ void k_mix(const float* __restrict__ x,
                      const float* __restrict__ mr, const float* __restrict__ mw,
                      const float* __restrict__ mk, const float* __restrict__ mv,
                      const float* __restrict__ ma, const float* __restrict__ mg,
                      unsigned short* __restrict__ xr, unsigned short* __restrict__ xw,
                      unsigned short* __restrict__ xk, unsigned short* __restrict__ xv,
                      unsigned short* __restrict__ xa, unsigned short* __restrict__ xg) {
  int i = blockIdx.x * blockDim.x + threadIdx.x;
  if (i >= Mn * Cn) return;
  int c  = i & (Cn - 1);
  int bt = i >> 10;               // Cn == 1024
  int t  = bt & (Tn - 1);
  float xc = x[i];
  float xp = (t == 0) ? 0.f : x[i - Cn];
  float xx = xp - xc;
  xr[i] = f2bf(xc + xx * mr[c]);
  xw[i] = f2bf(xc + xx * mw[c]);
  xk[i] = f2bf(xc + xx * mk[c]);
  xv[i] = f2bf(xc + xx * mv[c]);
  xa[i] = f2bf(xc + xx * ma[c]);
  xg[i] = f2bf(xc + xx * mg[c]);
}

// ---------------- WMMA bf16 GEMM: C[M,N] = A[M,K] @ Bt[N,K]^T (+ addend) ---------
// One wave per 64x32 tile: 4 A-frags x 2 B-frags -> 8 v_wmma per 32-deep K step.
template <bool ADD>
__global__ void k_gemm_bf16(const unsigned short* __restrict__ A,
                            const unsigned short* __restrict__ Bt,
                            float* __restrict__ C,
                            const float* __restrict__ addend,
                            int Kdim, int Ndim) {
  const int tileN = blockIdx.x * 32;
  const int tileM = blockIdx.y * 64;
  const int lane  = threadIdx.x;          // 0..31, full wave (EXEC all ones)
  const int half  = lane >> 4;            // 0 or 1
  const int l16   = lane & 15;

  const unsigned short* ar[4];
#pragma unroll
  for (int i = 0; i < 4; ++i) ar[i] = A + (size_t)(tileM + i * 16 + l16) * Kdim;
  const unsigned short* br[2];
#pragma unroll
  for (int i = 0; i < 2; ++i) br[i] = Bt + (size_t)(tileN + i * 16 + l16) * Kdim;

  v8f acc[4][2] = {};

#pragma unroll 2
  for (int k0 = 0; k0 < Kdim; k0 += 32) {
    const int o1 = k0 + half * 8;
    const int o2 = k0 + 16 + half * 8;
    // speculative gfx1250 global_prefetch, 2 K-steps ahead (OOB silently dropped)
    __builtin_prefetch(ar[0] + k0 + 64, 0, 3);
    __builtin_prefetch(ar[1] + k0 + 64, 0, 3);
    __builtin_prefetch(ar[2] + k0 + 64, 0, 3);
    __builtin_prefetch(ar[3] + k0 + 64, 0, 3);
    __builtin_prefetch(br[0] + k0 + 64, 0, 3);
    __builtin_prefetch(br[1] + k0 + 64, 0, 3);
    v16bf b0 = ldfrag(br[0], o1, o2);
    v16bf b1 = ldfrag(br[1], o1, o2);
#pragma unroll
    for (int mi = 0; mi < 4; ++mi) {
      v16bf am = ldfrag(ar[mi], o1, o2);
      acc[mi][0] = __builtin_amdgcn_wmma_f32_16x16x32_bf16(false, am, false, b0,
                                                           (short)0, acc[mi][0], false, false);
      acc[mi][1] = __builtin_amdgcn_wmma_f32_16x16x32_bf16(false, am, false, b1,
                                                           (short)0, acc[mi][1], false, false);
    }
  }

  // D layout: VGPR r -> M = r (lanes 0-15) / M = 8+r (lanes 16-31); N = lane&15
#pragma unroll
  for (int mi = 0; mi < 4; ++mi) {
    const int rowbase = tileM + mi * 16 + half * 8;
#pragma unroll
    for (int ni = 0; ni < 2; ++ni) {
      const int col = tileN + ni * 16 + l16;
#pragma unroll
      for (int r = 0; r < 8; ++r) {
        size_t o = (size_t)(rowbase + r) * Ndim + col;
        float v = acc[mi][ni][r];
        if (ADD) v += addend[o];
        C[o] = v;
      }
    }
  }
}

// ---------------- fused prep: w/a/v/k/kk per (b,t,h) group ----------------
// NOTE: w_s stores the DECAY exp(log_w) directly (scan needs only that).
__global__ void k_prep(const float* __restrict__ k_raw, const float* __restrict__ v_raw,
                       const float* __restrict__ v_first,
                       const float* __restrict__ wl, const float* __restrict__ al,
                       const float* __restrict__ vl,
                       const float* __restrict__ w0, const float* __restrict__ a0,
                       const float* __restrict__ v0,
                       const float* __restrict__ k_k, const float* __restrict__ k_a,
                       float* __restrict__ w_s, float* __restrict__ k_s,
                       float* __restrict__ v_s, float* __restrict__ a_s,
                       float* __restrict__ b_s) {
  int gid  = blockIdx.x * (blockDim.x >> 5) + (threadIdx.x >> 5);
  int lane = threadIdx.x & 31;
  int h    = gid % Hn;
  size_t bt = (size_t)(gid / Hn);
  size_t base = bt * Cn + (size_t)h * Nh;
  int c0 = h * Nh + lane, c1 = c0 + 32;
  size_t i0 = base + lane, i1 = i0 + 32;

  float kr0 = k_raw[i0], kr1 = k_raw[i1];
  float kk0 = kr0 * k_k[c0], kk1 = kr1 * k_k[c1];
  float ss = kk0 * kk0 + kk1 * kk1;
#pragma unroll
  for (int m = 16; m; m >>= 1) ss += __shfl_xor(ss, m, 32);
  float inv_norm = 1.f / fmaxf(sqrtf(ss), 1e-12f);

#pragma unroll
  for (int halfc = 0; halfc < 2; ++halfc) {
    size_t i = halfc ? i1 : i0;
    int    c = halfc ? c1 : c0;
    float kr = halfc ? kr1 : kr0;
    float kkn = (halfc ? kk1 : kk0) * inv_norm;

    float a = sigmoidf_(a0[c] + al[i]);
    float z = w0[c] + wl[i];
    float sp = fmaxf(-z, 0.f) + log1pf(__expf(-fabsf(z)));  // softplus(-z)
    float logw = -__expf(-sp - 0.5f);
    float kmod = kr * (1.f + (a - 1.f) * k_a[c]);
    float vv = v_raw[i];
    vv = vv + (v_first[i] - vv) * sigmoidf_(v0[c] + vl[i]);

    w_s[i] = __expf(logw);      // decay factor, used directly by the scan
    k_s[i] = kmod;
    v_s[i] = vv;
    a_s[i] = -kkn;
    b_s[i] = kkn * a;
  }
}

// ---------------- sequential RWKV7 scan: one block (64 thr) per (b,h) -----------
// Double-buffered LDS pipeline: step t+1's 6 vectors are fetched (async-to-LDS on
// toolchains that expose it, else register-staged) while step t computes.
__global__ void k_scan(const float* __restrict__ rp, const float* __restrict__ wp,
                       const float* __restrict__ kp, const float* __restrict__ vp,
                       const float* __restrict__ ap, const float* __restrict__ bp,
                       float* __restrict__ o) {
  int bh = blockIdx.x;
  int bi = bh / Hn, h = bh % Hn;
  int tid = threadIdx.x;                    // state row index v
  __shared__ float sbuf[2][6][Nh];          // [buffer][r,w,k,v,a,b][n]

  const float* src[6] = {rp, wp, kp, vp, ap, bp};

  float S[Nh];
#pragma unroll
  for (int j = 0; j < Nh; ++j) S[j] = 0.f;

  size_t base = (size_t)bi * Tn * Cn + (size_t)h * Nh;

  // preload t=0 into buffer 0
  {
    size_t off = base + tid;
#if HAVE_ASYNC_LDS
#pragma unroll
    for (int s = 0; s < 6; ++s) async_ld(src[s] + off, &sbuf[0][s][tid]);
    __builtin_amdgcn_s_wait_asynccnt(0);
#else
#pragma unroll
    for (int s = 0; s < 6; ++s) sbuf[0][s][tid] = src[s][off];
#endif
  }
  __syncthreads();

  for (int t = 0; t < Tn; ++t) {
    const int cur = t & 1, nxt = cur ^ 1;
    const size_t offn = base + (size_t)(t + 1) * Cn + tid;

#if HAVE_ASYNC_LDS
    if (t + 1 < Tn) {
#pragma unroll
      for (int s = 0; s < 6; ++s) async_ld(src[s] + offn, &sbuf[nxt][s][tid]);
    }
#else
    float pre[6];
    if (t + 1 < Tn) {
#pragma unroll
      for (int s = 0; s < 6; ++s) pre[s] = src[s][offn];
    }
#endif

    const float* sr = sbuf[cur][0];
    const float* sw = sbuf[cur][1];
    const float* sk = sbuf[cur][2];
    const float* sv = sbuf[cur][3];
    const float* sa = sbuf[cur][4];
    const float* sb = sbuf[cur][5];

    float sa_dot = 0.f;
#pragma unroll
    for (int j = 0; j < Nh; ++j) sa_dot += S[j] * sa[j];

    float vv = sv[tid];
    float od = 0.f;
#pragma unroll
    for (int j = 0; j < Nh; ++j) {
      float s = S[j] * sw[j] + sa_dot * sb[j] + vv * sk[j];
      S[j] = s;
      od += s * sr[j];
    }
    o[base + (size_t)t * Cn + tid] = od;

#if HAVE_ASYNC_LDS
    __builtin_amdgcn_s_wait_asynccnt(0);
#else
    if (t + 1 < Tn) {
#pragma unroll
      for (int s = 0; s < 6; ++s) sbuf[nxt][s][tid] = pre[s];
    }
#endif
    __syncthreads();
  }
}

// ---------------- fused GroupNorm + r.k.r_k bonus + gate -> bf16 ----------------
__global__ void k_post(const float* __restrict__ o, const float* __restrict__ r4,
                       const float* __restrict__ k4, const float* __restrict__ v4,
                       const float* __restrict__ g, const float* __restrict__ r_k,
                       const float* __restrict__ ln_w, const float* __restrict__ ln_b,
                       unsigned short* __restrict__ yg) {
  int gid  = blockIdx.x * (blockDim.x >> 5) + (threadIdx.x >> 5);
  int lane = threadIdx.x & 31;
  int h    = gid % Hn;
  size_t bt = (size_t)(gid / Hn);
  size_t base = bt * Cn + (size_t)h * Nh;
  size_t i0 = base + lane, i1 = i0 + 32;
  int c0 = h * Nh + lane, c1 = c0 + 32;

  float o0 = o[i0], o1 = o[i1];
  float s = o0 + o1;
#pragma unroll
  for (int m = 16; m; m >>= 1) s += __shfl_xor(s, m, 32);
  float mu = s * (1.f / Nh);
  float d0 = o0 - mu, d1 = o1 - mu;
  float vs = d0 * d0 + d1 * d1;
#pragma unroll
  for (int m = 16; m; m >>= 1) vs += __shfl_xor(vs, m, 32);
  float rstd = rsqrtf(vs * (1.f / Nh) + 0.00064f);

  float rk = r4[i0] * k4[i0] * r_k[c0] + r4[i1] * k4[i1] * r_k[c1];
#pragma unroll
  for (int m = 16; m; m >>= 1) rk += __shfl_xor(rk, m, 32);

  float y0 = d0 * rstd * ln_w[c0] + ln_b[c0] + rk * v4[i0];
  float y1 = d1 * rstd * ln_w[c1] + ln_b[c1] + rk * v4[i1];
  yg[i0] = f2bf(y0 * g[i0]);
  yg[i1] = f2bf(y1 * g[i1]);
}

// =====================================================================
extern "C" void kernel_launch(void* const* d_in, const int* in_sizes, int n_in,
                              void* d_out, int out_size, void* d_ws, size_t ws_size,
                              hipStream_t stream) {
  (void)in_sizes; (void)n_in; (void)out_size; (void)ws_size;
  const float* residual = (const float*)d_in[0];
  const float* x        = (const float*)d_in[1];
  const float* v_first  = (const float*)d_in[2];
  // d_in[3]=x0, d_in[4]=dx0, d_in[5]=token_ids : unused by reference
  const float* mr = (const float*)d_in[6];
  const float* mw = (const float*)d_in[7];
  const float* mk = (const float*)d_in[8];
  const float* mv = (const float*)d_in[9];
  const float* ma = (const float*)d_in[10];
  const float* mg = (const float*)d_in[11];
  const float* w0 = (const float*)d_in[12];
  const float* w1 = (const float*)d_in[13];
  const float* w2 = (const float*)d_in[14];
  const float* a0 = (const float*)d_in[15];
  const float* a1 = (const float*)d_in[16];
  const float* a2 = (const float*)d_in[17];
  const float* v0 = (const float*)d_in[18];
  const float* v1 = (const float*)d_in[19];
  const float* v2 = (const float*)d_in[20];
  const float* g1 = (const float*)d_in[21];
  const float* g2 = (const float*)d_in[22];
  const float* k_k = (const float*)d_in[23];
  const float* k_a = (const float*)d_in[24];
  const float* r_k = (const float*)d_in[25];
  const float* W_r = (const float*)d_in[26];
  const float* W_k = (const float*)d_in[27];
  const float* W_v = (const float*)d_in[28];
  const float* W_o = (const float*)d_in[29];
  const float* ln_w = (const float*)d_in[30];
  const float* ln_b = (const float*)d_in[31];
  float* out = (float*)d_out;

  // ---- workspace bump allocator ----
  size_t off = 0;
  char* wsb = (char*)d_ws;
  auto A256 = [&](size_t bytes) -> char* {
    char* p = wsb + off;
    off += (bytes + 255) & ~(size_t)255;
    return p;
  };
  typedef unsigned short u16;
  // bf16 weights (transposed to [N,K] for WMMA B operands)
  u16* Wr_bt = (u16*)A256((size_t)Cn * Cn * 2);
  u16* Wk_bt = (u16*)A256((size_t)Cn * Cn * 2);
  u16* Wv_bt = (u16*)A256((size_t)Cn * Cn * 2);
  u16* Wo_bt = (u16*)A256((size_t)Cn * Cn * 2);
  u16* w1_bt = (u16*)A256((size_t)Cn * DW * 2);
  u16* w2_bt = (u16*)A256((size_t)DW * Cn * 2);
  u16* a1_bt = (u16*)A256((size_t)Cn * DA * 2);
  u16* a2_bt = (u16*)A256((size_t)DA * Cn * 2);
  u16* v1_bt = (u16*)A256((size_t)Cn * DV * 2);
  u16* v2_bt = (u16*)A256((size_t)DV * Cn * 2);
  u16* g1_bt = (u16*)A256((size_t)Cn * DG * 2);
  u16* g2_bt = (u16*)A256((size_t)DG * Cn * 2);
  // bf16 activations (A operands, row-major)
  u16* xr_bf = (u16*)A256((size_t)Mn * Cn * 2);
  u16* xw_bf = (u16*)A256((size_t)Mn * Cn * 2);
  u16* xk_bf = (u16*)A256((size_t)Mn * Cn * 2);
  u16* xv_bf = (u16*)A256((size_t)Mn * Cn * 2);
  u16* xa_bf = (u16*)A256((size_t)Mn * Cn * 2);
  u16* xg_bf = (u16*)A256((size_t)Mn * Cn * 2);
  u16* hw_bf = (u16*)A256((size_t)Mn * DW * 2);
  u16* ha_bf = (u16*)A256((size_t)Mn * DA * 2);
  u16* hv_bf = (u16*)A256((size_t)Mn * DV * 2);
  u16* hg_bf = (u16*)A256((size_t)Mn * DG * 2);
  u16* yg_bf = (u16*)A256((size_t)Mn * Cn * 2);
  // f32 intermediates
  float* r_f  = (float*)A256((size_t)Mn * Cn * 4);
  float* k_f  = (float*)A256((size_t)Mn * Cn * 4);
  float* v_f  = (float*)A256((size_t)Mn * Cn * 4);
  float* wl_f = (float*)A256((size_t)Mn * Cn * 4);
  float* al_f = (float*)A256((size_t)Mn * Cn * 4);
  float* vl_f = (float*)A256((size_t)Mn * Cn * 4);
  float* g_f  = (float*)A256((size_t)Mn * Cn * 4);
  float* hw_f = (float*)A256((size_t)Mn * DW * 4);
  float* ha_f = (float*)A256((size_t)Mn * DA * 4);
  float* hv_f = (float*)A256((size_t)Mn * DV * 4);
  float* hg_f = (float*)A256((size_t)Mn * DG * 4);
  float* w_s  = (float*)A256((size_t)Mn * Cn * 4);
  float* k_s  = (float*)A256((size_t)Mn * Cn * 4);
  float* v_s  = (float*)A256((size_t)Mn * Cn * 4);
  float* a_s  = (float*)A256((size_t)Mn * Cn * 4);
  float* b_s  = (float*)A256((size_t)Mn * Cn * 4);
  float* o_f  = (float*)A256((size_t)Mn * Cn * 4);

  const int CT = 256;
  auto cvt_t = [&](const float* s, u16* d, int K, int N) {
    k_f2bf_t<<<(K * N + CT - 1) / CT, CT, 0, stream>>>(s, d, K, N);
  };
  // 1) weights -> bf16, transposed to [N,K]
  cvt_t(W_r, Wr_bt, Cn, Cn);  cvt_t(W_k, Wk_bt, Cn, Cn);
  cvt_t(W_v, Wv_bt, Cn, Cn);  cvt_t(W_o, Wo_bt, Cn, Cn);
  cvt_t(w1, w1_bt, Cn, DW);   cvt_t(w2, w2_bt, DW, Cn);
  cvt_t(a1, a1_bt, Cn, DA);   cvt_t(a2, a2_bt, DA, Cn);
  cvt_t(v1, v1_bt, Cn, DV);   cvt_t(v2, v2_bt, DV, Cn);
  cvt_t(g1, g1_bt, Cn, DG);   cvt_t(g2, g2_bt, DG, Cn);

  // 2) token shift + mix
  k_mix<<<(Mn * Cn + CT - 1) / CT, CT, 0, stream>>>(
      x, mr, mw, mk, mv, ma, mg, xr_bf, xw_bf, xk_bf, xv_bf, xa_bf, xg_bf);

  auto gemm = [&](const u16* Am, const u16* Bt, float* Cm, const float* add,
                  int Md, int Kd, int Nd) {
    dim3 grid(Nd / 32, Md / 64);
    if (add)
      k_gemm_bf16<true><<<grid, 32, 0, stream>>>(Am, Bt, Cm, add, Kd, Nd);
    else
      k_gemm_bf16<false><<<grid, 32, 0, stream>>>(Am, Bt, Cm, nullptr, Kd, Nd);
  };

  // 3) big projections (WMMA)
  gemm(xr_bf, Wr_bt, r_f, nullptr, Mn, Cn, Cn);
  gemm(xk_bf, Wk_bt, k_f, nullptr, Mn, Cn, Cn);
  gemm(xv_bf, Wv_bt, v_f, nullptr, Mn, Cn, Cn);

  // 4) LoRA pipelines (WMMA)
  gemm(xw_bf, w1_bt, hw_f, nullptr, Mn, Cn, DW);
  k_tanh_bf<<<(Mn * DW + CT - 1) / CT, CT, 0, stream>>>(hw_f, hw_bf, Mn * DW);
  gemm(hw_bf, w2_bt, wl_f, nullptr, Mn, DW, Cn);

  gemm(xa_bf, a1_bt, ha_f, nullptr, Mn, Cn, DA);
  k_f2bf<<<(Mn * DA + CT - 1) / CT, CT, 0, stream>>>(ha_f, ha_bf, Mn * DA);
  gemm(ha_bf, a2_bt, al_f, nullptr, Mn, DA, Cn);

  gemm(xv_bf, v1_bt, hv_f, nullptr, Mn, Cn, DV);
  k_f2bf<<<(Mn * DV + CT - 1) / CT, CT, 0, stream>>>(hv_f, hv_bf, Mn * DV);
  gemm(hv_bf, v2_bt, vl_f, nullptr, Mn, DV, Cn);

  gemm(xg_bf, g1_bt, hg_f, nullptr, Mn, Cn, DG);
  k_sig_bf<<<(Mn * DG + CT - 1) / CT, CT, 0, stream>>>(hg_f, hg_bf, Mn * DG);
  gemm(hg_bf, g2_bt, g_f, nullptr, Mn, DG, Cn);

  // 5) fused prep per (b,t,h)
  {
    int groups = Bn * Tn * Hn;               // 32768
    int wavesPerBlock = CT / 32;             // 8
    k_prep<<<groups / wavesPerBlock, CT, 0, stream>>>(
        k_f, v_f, v_first, wl_f, al_f, vl_f, w0, a0, v0, k_k, k_a,
        w_s, k_s, v_s, a_s, b_s);
  }

  // 6) sequential scan (double-buffered LDS pipeline)
  k_scan<<<Bn * Hn, Nh, 0, stream>>>(r_f, w_s, k_s, v_s, a_s, b_s, o_f);

  // 7) groupnorm + bonus + gate -> bf16
  {
    int groups = Bn * Tn * Hn;
    int wavesPerBlock = CT / 32;
    k_post<<<groups / wavesPerBlock, CT, 0, stream>>>(
        o_f, r_f, k_s, v_s, g_f, r_k, ln_w, ln_b, yg_bf);
  }

  // 8) output projection + residual (WMMA, fused add)
  gemm(yg_bf, Wo_bt, out, residual, Mn, Cn, Cn);
}
